// SequentialTransductionUnit_1786706395701
// MI455X (gfx1250) — matrix operand order repaired and checked by reading, not compiled, not run on hardware
//
#include <hip/hip_runtime.h>
#include <hip/hip_bf16.h>
#include <math.h>

#define D_    1024
#define H_    8
#define L_    2048
#define B_    2
#define HD_   128
#define M_TOT (B_ * L_)   // 4096
#define NBUCK 128
#define EPS_  1e-6f

typedef __attribute__((ext_vector_type(16))) __bf16 v16bf;
typedef __attribute__((ext_vector_type(8)))  __bf16 v8bf;
typedef __attribute__((ext_vector_type(8)))  float  v8f;

__device__ __forceinline__ float silu_f(float x) {
    return x / (1.0f + __expf(-x));
}

// A-operand fragment (16-bit A, 16x32): per lane, elements 0..7 at base,
// elements 8..15 at base+8 (lane already pre-offset by hi*16 in K).
__device__ __forceinline__ v16bf load_a_frag(const __bf16* base) {
    v8bf lo = *(const v8bf*)base;
    v8bf hi = *(const v8bf*)(base + 8);
    v16bf a;
#pragma unroll
    for (int i = 0; i < 8; ++i) { a[i] = lo[i]; a[8 + i] = hi[i]; }
    return a;
}

__device__ __forceinline__ v8f wmma_bf16(v16bf a, v16bf b, v8f c) {
    return __builtin_amdgcn_wmma_f32_16x16x32_bf16(
        /*neg_a=*/false, a, /*neg_b=*/false, b,
        /*c_mod=*/(short)0, c, /*reuse_a=*/false, /*reuse_b=*/false);
}

// ---------------------------------------------------------------------------
// K0: convert the five DxD f32 weights to bf16 (packed consecutively).
// ---------------------------------------------------------------------------
__global__ void k_cvt_weights(const float* __restrict__ wq, const float* __restrict__ wk,
                              const float* __restrict__ wv, const float* __restrict__ wu,
                              const float* __restrict__ wo, __bf16* __restrict__ out) {
    int i = blockIdx.x * blockDim.x + threadIdx.x;
    int w = i >> 20;
    int j = i & ((1 << 20) - 1);
    float v;
    if (w == 0)      v = wq[j];
    else if (w == 1) v = wk[j];
    else if (w == 2) v = wv[j];
    else if (w == 3) v = wu[j];
    else             v = wo[j];
    out[i] = (__bf16)v;
}

// ---------------------------------------------------------------------------
// K1: h = rmsnorm(hidden, ln1_w) -> bf16.  One block (256 thr) per row.
// ---------------------------------------------------------------------------
__global__ void k_rmsnorm1(const float* __restrict__ x, const float* __restrict__ w,
                           __bf16* __restrict__ hbf) {
    int row = blockIdx.x;
    int t   = threadIdx.x;
    float4 v = ((const float4*)(x + (size_t)row * D_))[t];
    float ss = v.x * v.x + v.y * v.y + v.z * v.z + v.w * v.w;
#pragma unroll
    for (int off = 16; off > 0; off >>= 1) ss += __shfl_xor(ss, off, 32);
    __shared__ float red[8];
    if ((t & 31) == 0) red[t >> 5] = ss;
    __syncthreads();
    float tot = 0.f;
#pragma unroll
    for (int i = 0; i < 8; ++i) tot += red[i];
    float scale = rsqrtf(tot * (1.0f / D_) + EPS_);
    float4 wv4 = ((const float4*)w)[t];
    __bf16* o = hbf + (size_t)row * D_ + t * 4;
    o[0] = (__bf16)(v.x * scale * wv4.x);
    o[1] = (__bf16)(v.y * scale * wv4.y);
    o[2] = (__bf16)(v.z * scale * wv4.z);
    o[3] = (__bf16)(v.w * scale * wv4.w);
}

// ---------------------------------------------------------------------------
// K2: projections: silu(h @ W^T) for q,k,v,u (mode = blockIdx.z).
//   q,k -> bf16 (B,H,L,hd);  v -> bf16 transposed (B,H,hd,L);  u -> f32.
//   Block = 8 waves; each wave owns a 16x64 output tile (4 accumulators).
// ---------------------------------------------------------------------------
__global__ void k_proj_gemm(const __bf16* __restrict__ hbf, const __bf16* __restrict__ wbf,
                            __bf16* __restrict__ qbf, __bf16* __restrict__ kbf,
                            __bf16* __restrict__ vtbf, float* __restrict__ uf) {
    const int mode = blockIdx.z;
    const __bf16* W = wbf + (size_t)mode * (D_ * D_);
    const int lane = threadIdx.x & 31;
    const int wave = threadIdx.x >> 5;
    const int ln = lane & 15;
    const int hi = lane >> 4;
    const int m0 = blockIdx.x * 128 + wave * 16;
    const int n0 = blockIdx.y * 64;

    v8f zero = {};
    v8f acc[4];
#pragma unroll
    for (int i = 0; i < 4; ++i) acc[i] = zero;

    const __bf16* arow = hbf + (size_t)(m0 + ln) * D_ + hi * 16;
    for (int k0 = 0; k0 < D_; k0 += 32) {
        __builtin_prefetch(arow + k0 + 128, 0, 0);
        v16bf a = load_a_frag(arow + k0);
#pragma unroll
        for (int nf = 0; nf < 4; ++nf) {
            const __bf16* brow = W + (size_t)(n0 + nf * 16 + ln) * D_ + k0 + hi * 16;
            v16bf b = *(const v16bf*)brow;
            acc[nf] = wmma_bf16(a, b, acc[nf]);
        }
    }

#pragma unroll
    for (int nf = 0; nf < 4; ++nf) {
#pragma unroll
        for (int r = 0; r < 8; ++r) {
            int m = m0 + r + hi * 8;
            int n = n0 + nf * 16 + ln;
            float y = silu_f(acc[nf][r]);
            int b = m >> 11;            // / L
            int l = m & (L_ - 1);
            int head = n >> 7;
            int d = n & (HD_ - 1);
            size_t bh = (size_t)(b * H_ + head);
            if (mode == 0)      qbf[(bh * L_ + l) * HD_ + d] = (__bf16)y;
            else if (mode == 1) kbf[(bh * L_ + l) * HD_ + d] = (__bf16)y;
            else if (mode == 2) vtbf[(bh * HD_ + d) * L_ + l] = (__bf16)y;
            else                uf[(size_t)m * D_ + n] = y;
        }
    }
}

// ---------------------------------------------------------------------------
// K3: qk = silu(q @ k^T + pos_bias + ts_bias)/L * mask -> f32 to d_out.
//   Wave owns a 16-row strip, loops all 2048 columns in 16-wide tiles.
//   q strip A-frags (K=128 -> 4 frags) are kept in registers for the loop.
// ---------------------------------------------------------------------------
__global__ void k_qk(const __bf16* __restrict__ qbf, const __bf16* __restrict__ kbf,
                     const int* __restrict__ ts, const unsigned char* __restrict__ msk,
                     const float* __restrict__ posw, const float* __restrict__ tsw,
                     float* __restrict__ qk_out) {
    const int bh = blockIdx.y;
    const int b = bh >> 3;
    const int lane = threadIdx.x & 31;
    const int wave = threadIdx.x >> 5;
    const int ln = lane & 15;
    const int hi = lane >> 4;
    const int m0 = blockIdx.x * 128 + wave * 16;

    const __bf16* qrow = qbf + ((size_t)bh * L_ + (m0 + ln)) * HD_ + hi * 16;
    v16bf aq[4];
#pragma unroll
    for (int kk = 0; kk < 4; ++kk) aq[kk] = load_a_frag(qrow + kk * 32);

    // ext[m+1] per output row of this lane
    int tsm[8];
#pragma unroll
    for (int r = 0; r < 8; ++r) {
        int m = m0 + r + hi * 8;
        int mp = m + 1; if (mp > L_ - 1) mp = L_ - 1;
        tsm[r] = ts[b * L_ + mp];
    }

    const float inv_log = 1.0f / 0.301f;
    v8f zero = {};
    for (int j = 0; j < L_; j += 16) {
        const __bf16* krow = kbf + ((size_t)bh * L_ + (j + ln)) * HD_ + hi * 16;
        v8f acc = zero;
#pragma unroll
        for (int kk = 0; kk < 4; ++kk) {
            v16bf bfrag = *(const v16bf*)(krow + kk * 32);
            acc = wmma_bf16(aq[kk], bfrag, acc);
        }
        int n = j + ln;
        int tsn = ts[b * L_ + n];
#pragma unroll
        for (int r = 0; r < 8; ++r) {
            int m = m0 + r + hi * 8;
            float x = acc[r];
            int diff = tsm[r] - tsn;
            int ad = diff < 0 ? -diff : diff;
            if (ad < 1) ad = 1;
            int bkt = (int)(__logf((float)ad) * inv_log);
            bkt = bkt < 0 ? 0 : (bkt > NBUCK ? NBUCK : bkt);
            float bias = posw[n - m + (L_ - 1)] + tsw[bkt];
            x = silu_f(x + bias) * (1.0f / L_);
            x *= (float)msk[((size_t)b * L_ + m) * L_ + n];
            qk_out[((size_t)bh * L_ + m) * L_ + n] = x;
        }
    }
}

// ---------------------------------------------------------------------------
// K4: attn = qk @ v.  qk read back from d_out (f32) and converted to bf16
//   in-register for the A operand; v read from the transposed bf16 copy so
//   B-frag columns are contiguous along K.  Wave owns 16 rows x 128 cols.
// ---------------------------------------------------------------------------
__global__ void k_attn(const float* __restrict__ qk, const __bf16* __restrict__ vtbf,
                       float* __restrict__ attn) {
    const int bh = blockIdx.y;
    const int b = bh >> 3;
    const int h = bh & 7;
    const int lane = threadIdx.x & 31;
    const int wave = threadIdx.x >> 5;
    const int ln = lane & 15;
    const int hi = lane >> 4;
    const int m0 = (blockIdx.x * 8 + wave) * 16;

    v8f zero = {};
    v8f acc[8];
#pragma unroll
    for (int i = 0; i < 8; ++i) acc[i] = zero;

    const float*  arow  = qk + ((size_t)bh * L_ + (m0 + ln)) * L_ + hi * 16;
    const __bf16* vbase = vtbf + (size_t)bh * HD_ * L_ + hi * 16;

    for (int k0 = 0; k0 < L_; k0 += 32) {
        __builtin_prefetch(arow + k0 + 64, 0, 0);
        const float4* p = (const float4*)(arow + k0);
        float4 f0 = p[0], f1 = p[1];   // K = k0+kb+0..7
        float4 f2 = p[2], f3 = p[3];   // K = k0+kb+8..15
        v16bf a;
        a[0]  = (__bf16)f0.x; a[1]  = (__bf16)f0.y; a[2]  = (__bf16)f0.z; a[3]  = (__bf16)f0.w;
        a[4]  = (__bf16)f1.x; a[5]  = (__bf16)f1.y; a[6]  = (__bf16)f1.z; a[7]  = (__bf16)f1.w;
        a[8]  = (__bf16)f2.x; a[9]  = (__bf16)f2.y; a[10] = (__bf16)f2.z; a[11] = (__bf16)f2.w;
        a[12] = (__bf16)f3.x; a[13] = (__bf16)f3.y; a[14] = (__bf16)f3.z; a[15] = (__bf16)f3.w;
#pragma unroll
        for (int nf = 0; nf < 8; ++nf) {
            const __bf16* brow = vbase + (size_t)(nf * 16 + ln) * L_ + k0;
            v16bf bf = *(const v16bf*)brow;
            acc[nf] = wmma_bf16(a, bf, acc[nf]);
        }
    }
#pragma unroll
    for (int nf = 0; nf < 8; ++nf) {
#pragma unroll
        for (int r = 0; r < 8; ++r) {
            int l = m0 + r + hi * 8;
            int d = nf * 16 + ln;
            attn[((size_t)(b * L_ + l)) * D_ + h * HD_ + d] = acc[nf][r];
        }
    }
}

// ---------------------------------------------------------------------------
// K5: t = u * rmsnorm(attn, ln2_w) -> bf16.
// ---------------------------------------------------------------------------
__global__ void k_rmsnorm2(const float* __restrict__ attn, const float* __restrict__ u,
                           const float* __restrict__ w, __bf16* __restrict__ tbf) {
    int row = blockIdx.x;
    int t   = threadIdx.x;
    float4 v = ((const float4*)(attn + (size_t)row * D_))[t];
    float ss = v.x * v.x + v.y * v.y + v.z * v.z + v.w * v.w;
#pragma unroll
    for (int off = 16; off > 0; off >>= 1) ss += __shfl_xor(ss, off, 32);
    __shared__ float red[8];
    if ((t & 31) == 0) red[t >> 5] = ss;
    __syncthreads();
    float tot = 0.f;
#pragma unroll
    for (int i = 0; i < 8; ++i) tot += red[i];
    float scale = rsqrtf(tot * (1.0f / D_) + EPS_);
    float4 wv4 = ((const float4*)w)[t];
    float4 uv  = ((const float4*)(u + (size_t)row * D_))[t];
    __bf16* o = tbf + (size_t)row * D_ + t * 4;
    o[0] = (__bf16)(uv.x * v.x * scale * wv4.x);
    o[1] = (__bf16)(uv.y * v.y * scale * wv4.y);
    o[2] = (__bf16)(uv.z * v.z * scale * wv4.z);
    o[3] = (__bf16)(uv.w * v.w * scale * wv4.w);
}

// ---------------------------------------------------------------------------
// K6: out0 = residual + t @ w_o^T.
// ---------------------------------------------------------------------------
__global__ void k_out_gemm(const __bf16* __restrict__ tbf, const __bf16* __restrict__ wobf,
                           const float* __restrict__ residual, float* __restrict__ out0) {
    const int lane = threadIdx.x & 31;
    const int wave = threadIdx.x >> 5;
    const int ln = lane & 15;
    const int hi = lane >> 4;
    const int m0 = blockIdx.x * 128 + wave * 16;
    const int n0 = blockIdx.y * 64;

    v8f zero = {};
    v8f acc[4];
#pragma unroll
    for (int i = 0; i < 4; ++i) acc[i] = zero;

    const __bf16* arow = tbf + (size_t)(m0 + ln) * D_ + hi * 16;
    for (int k0 = 0; k0 < D_; k0 += 32) {
        v16bf a = load_a_frag(arow + k0);
#pragma unroll
        for (int nf = 0; nf < 4; ++nf) {
            const __bf16* brow = wobf + (size_t)(n0 + nf * 16 + ln) * D_ + k0 + hi * 16;
            acc[nf] = wmma_bf16(a, *(const v16bf*)brow, acc[nf]);
        }
    }
#pragma unroll
    for (int nf = 0; nf < 4; ++nf) {
#pragma unroll
        for (int r = 0; r < 8; ++r) {
            int m = m0 + r + hi * 8;
            int n = n0 + nf * 16 + ln;
            out0[(size_t)m * D_ + n] = residual[(size_t)m * D_ + n] + acc[nf][r];
        }
    }
}

// ---------------------------------------------------------------------------
extern "C" void kernel_launch(void* const* d_in, const int* in_sizes, int n_in,
                              void* d_out, int out_size, void* d_ws, size_t ws_size,
                              hipStream_t stream) {
    const float*         hidden = (const float*)d_in[0];
    const unsigned char* msk    = (const unsigned char*)d_in[1];
    const int*           ts     = (const int*)d_in[2];
    const float*         wq     = (const float*)d_in[3];
    const float*         wk     = (const float*)d_in[4];
    const float*         wv     = (const float*)d_in[5];
    const float*         wu     = (const float*)d_in[6];
    const float*         wo     = (const float*)d_in[7];
    const float*         ln1    = (const float*)d_in[8];
    const float*         ln2    = (const float*)d_in[9];
    const float*         posw   = (const float*)d_in[10];
    const float*         tsw    = (const float*)d_in[11];

    float* out0   = (float*)d_out;                 // (B,L,D)
    float* qk_out = out0 + (size_t)M_TOT * D_;     // (B,H,L,L)

    // workspace layout (bytes)
    char* ws = (char*)d_ws;
    __bf16* wbf  = (__bf16*)(ws);                  // 5 * 1M bf16   = 10,485,760 B
    __bf16* hbf  = (__bf16*)(ws + 10485760);       // 4M bf16       =  8,388,608 B
    __bf16* qbf  = (__bf16*)(ws + 18874368);       // 4M bf16
    __bf16* kbf  = (__bf16*)(ws + 27262976);       // 4M bf16
    __bf16* vtbf = (__bf16*)(ws + 35651584);       // 4M bf16
    float*  uf   = (float* )(ws + 44040192);       // 4M f32        = 16,777,216 B
    float*  attn = (float* )(ws + 60817408);       // 4M f32
    __bf16* tbf  = (__bf16*)(ws + 77594624);       // 4M bf16  -> total ~86 MB

    k_cvt_weights<<<(5 * 1048576) / 256, 256, 0, stream>>>(wq, wk, wv, wu, wo, wbf);
    k_rmsnorm1<<<M_TOT, 256, 0, stream>>>(hidden, ln1, hbf);
    k_proj_gemm<<<dim3(M_TOT / 128, D_ / 64, 4), 256, 0, stream>>>(hbf, wbf, qbf, kbf, vtbf, uf);
    k_qk<<<dim3(L_ / 128, B_ * H_), 256, 0, stream>>>(qbf, kbf, ts, msk, posw, tsw, qk_out);
    k_attn<<<dim3(L_ / 128, B_ * H_), 256, 0, stream>>>(qk_out, vtbf, attn);
    k_rmsnorm2<<<M_TOT, 256, 0, stream>>>(attn, uf, ln2, tbf);
    k_out_gemm<<<dim3(M_TOT / 128, D_ / 64), 256, 0, stream>>>(tbf, wbf + 4 * 1048576, hidden, out0);
}
